// LowRank3d_21062519619838
// MI455X (gfx1250) — matrix operand order repaired and checked by reading, not compiled, not run on hardware
//
#include <hip/hip_runtime.h>
#include <hip/hip_bf16.h>

// Problem constants (match reference)
#define S1 32
#define S2 32
#define S3 33
#define NX (S1 * S2 * S3)   // 33792 (divisible by 16 -> 2112 tiles)
#define BATCH 4
#define CIN 32
#define COUT 32

typedef __attribute__((ext_vector_type(16))) _Float16 v16h;
typedef __attribute__((ext_vector_type(8)))  _Float16 v8h;
typedef __attribute__((ext_vector_type(8)))  float    v8f;

__device__ __forceinline__ float elu(float x) {
    return x > 0.f ? x : (__expf(x) - 1.f);
}

// A-fragment (16x32 f16, MxK): lanes 0-15 hold row M=lane with K 0..15,
// lanes 16-31 hold row M=lane-16 with K 16..31.  p points at the lane's
// first K element; halves are 8 f16 apart.
__device__ __forceinline__ v16h load_afrag(const _Float16* p) {
    v8h lo = *(const v8h*)p;
    v8h hi = *(const v8h*)(p + 8);
    v16h a;
#pragma unroll
    for (int j = 0; j < 8; ++j) { a[j] = lo[j]; a[8 + j] = hi[j]; }
    return a;
}

// B-fragment (32x16 f16, KxN): lane holds column N=lane&15; lanes 0-15 hold
// K 0..15 contiguous, lanes 16-31 hold K 16..31 contiguous (same order as load_afrag,
// not the ISA table order).  With weights packed [n][k] this is one aligned 32B load.
__device__ __forceinline__ v16h load_bfrag(const _Float16* p) {
    return *(const v16h*)p;
}

__device__ __forceinline__ v8f wmma_f16(v16h a, v16h b, v8f c) {
    // v_wmma_f32_16x16x32_f16  D = A*B + C, f32 accumulate
    return __builtin_amdgcn_wmma_f32_16x16x32_f16(false, a, false, b, (short)0, c,
                                                  false, false);
}

// ---------------------------------------------------------------------------
// Kernel 0: pack w2 (64x128) and w3 (128x1024) into f16 [n][k] order; zero A.
// ---------------------------------------------------------------------------
__global__ void lr3d_pack(const float* __restrict__ w2phi, const float* __restrict__ w3phi,
                          const float* __restrict__ w2psi, const float* __restrict__ w3psi,
                          _Float16* __restrict__ w2p_phi, _Float16* __restrict__ w3p_phi,
                          _Float16* __restrict__ w2p_psi, _Float16* __restrict__ w3p_psi,
                          float* __restrict__ A_ws) {
    const int id     = blockIdx.x * blockDim.x + threadIdx.x;
    const int stride = gridDim.x * blockDim.x;
    for (int idx = id; idx < 128 * 1024; idx += stride) {
        int n = idx & 1023, k = idx >> 10;           // w3[k][n] -> w3p[n][k]
        w3p_phi[n * 128 + k] = (_Float16)w3phi[idx];
        w3p_psi[n * 128 + k] = (_Float16)w3psi[idx];
    }
    for (int idx = id; idx < 64 * 128; idx += stride) {
        int n = idx & 127, k = idx >> 7;             // w2[k][n] -> w2p[n][k]
        w2p_phi[n * 64 + k] = (_Float16)w2phi[idx];
        w2p_psi[n * 64 + k] = (_Float16)w2psi[idx];
    }
    for (int idx = id; idx < BATCH * COUT * CIN; idx += stride) A_ws[idx] = 0.f;
}

// ---------------------------------------------------------------------------
// Shared MLP front half: grid coords -> h1 (16x64, scalar) -> h2 (16x128, WMMA)
// h1/h2 kept in LDS as f16 tiles (WMMA A-operand staging).
// ---------------------------------------------------------------------------
__device__ __forceinline__ void mlp_front(int n0,
        const float* __restrict__ w1, const float* __restrict__ b1,
        const _Float16* __restrict__ w2p, const float* __restrict__ b2,
        _Float16 (*h1)[72], _Float16 (*h2)[136]) {
    const int tid  = threadIdx.x;
    const int lane = tid & 31;
    const int wave = tid >> 5;

    // Layer 1: 16x3 @ 3x64 — tiny, scalar f32, ELU, store f16 to LDS.
#pragma unroll
    for (int idx = tid; idx < 16 * 64; idx += 128) {
        int r = idx >> 6, c = idx & 63;
        int n  = n0 + r;
        int ix = n / (S2 * S3);
        int rem = n - ix * (S2 * S3);
        int iy = rem / S3;
        int it = rem - iy * S3;
        float gx = ix * (1.f / S1), gy = iy * (1.f / S2), gt = it * (1.f / (S3 - 1));
        float v = gx * w1[c] + gy * w1[64 + c] + gt * w1[128 + c] + b1[c];
        h1[r][c] = (_Float16)elu(v);
    }
    __syncthreads();

    // Layer 2: 16x64 @ 64x128 via WMMA (K=64 -> 2 steps, N=128 -> 8 tiles / 4 waves)
    const _Float16* arow = h1[lane & 15];
    const int koff = (lane & 16) ? 16 : 0;
    const int kb   = (lane & 16) ? 16 : 0;
    const int mrow = (lane & 16) ? 8 : 0;
    v16h a0 = load_afrag(arow + koff);
    v16h a1 = load_afrag(arow + 32 + koff);
    for (int nt = wave; nt < 8; nt += 4) {
        int n = nt * 16 + (lane & 15);
        v8f acc = {};
        acc = wmma_f16(a0, load_bfrag(w2p + n * 64 + kb), acc);
        acc = wmma_f16(a1, load_bfrag(w2p + n * 64 + 32 + kb), acc);
        float bias = b2[n];
#pragma unroll
        for (int r = 0; r < 8; ++r)
            h2[mrow + r][n] = (_Float16)elu(acc[r] + bias);
    }
    __syncthreads();
}

// ---------------------------------------------------------------------------
// Kernel 1: phi MLP tile (layer 3 via WMMA) fused with A[b,o,i] accumulation.
// ---------------------------------------------------------------------------
__global__ void __launch_bounds__(128, 2)
lr3d_phi_accum(const float* __restrict__ x,
               const float* __restrict__ w1, const float* __restrict__ b1,
               const _Float16* __restrict__ w2p, const float* __restrict__ b2,
               const _Float16* __restrict__ w3p, const float* __restrict__ b3,
               float* __restrict__ A_ws) {
    __shared__ __align__(16) _Float16 h1[16][72];
    __shared__ __align__(16) _Float16 h2[16][136];
    __shared__ __align__(16) float    xs[BATCH][CIN][16];

    const int tid  = threadIdx.x;
    const int lane = tid & 31;
    const int wave = tid >> 5;
    const int n0   = blockIdx.x * 16;

    // Stage this tile's x slice: xs[b][i][r] = x[b,i,n0+r]  (8 KB, coalesced)
    for (int idx = tid; idx < BATCH * CIN * 16; idx += 128) {
        int b = idx >> 9;
        int i = (idx >> 4) & 31;
        int r = idx & 15;
        xs[b][i][r] = x[((size_t)b * CIN + i) * NX + n0 + r];
    }

    mlp_front(n0, w1, b1, w2p, b2, h1, h2);

    // Layer 3: 16x128 @ 128x1024.  A-fragments loaded once, reused 64x.
    const _Float16* arow = h2[lane & 15];
    const int koff = (lane & 16) ? 16 : 0;
    const int kb   = (lane & 16) ? 16 : 0;
    const int mrow = (lane & 16) ? 8 : 0;
    v16h af[4];
#pragma unroll
    for (int kc = 0; kc < 4; ++kc) af[kc] = load_afrag(arow + kc * 32 + koff);

    for (int t = wave; t < 64; t += 4) {
        const int oi = t * 16 + (lane & 15);          // column = o*32 + i
        const _Float16* wp = w3p + oi * 128;
        v8f acc = {};
#pragma unroll
        for (int kc = 0; kc < 4; ++kc)
            acc = wmma_f16(af[kc], load_bfrag(wp + kc * 32 + kb), acc);
        if (t + 4 < 64) __builtin_prefetch(w3p + (oi + 64) * 128, 0, 1);

        const float bias = b3[oi];
        const int   i    = oi & 31;
#pragma unroll
        for (int b = 0; b < BATCH; ++b) {
            const float* xb = &xs[b][i][mrow];
            float4 x0 = *(const float4*)xb;
            float4 x1 = *(const float4*)(xb + 4);
            float p = (acc[0] + bias) * x0.x + (acc[1] + bias) * x0.y
                    + (acc[2] + bias) * x0.z + (acc[3] + bias) * x0.w
                    + (acc[4] + bias) * x1.x + (acc[5] + bias) * x1.y
                    + (acc[6] + bias) * x1.z + (acc[7] + bias) * x1.w;
            p += __shfl_xor(p, 16, 32);               // join the two half-rows
            if (!(lane & 16))
                atomicAdd(&A_ws[b * (COUT * CIN) + oi], p);
        }
    }
}

// ---------------------------------------------------------------------------
// Kernel 2: psi MLP tile (layer 3 via WMMA) fused with out[b,o,m] contraction.
// ---------------------------------------------------------------------------
__global__ void __launch_bounds__(128, 2)
lr3d_psi_out(const float* __restrict__ w1, const float* __restrict__ b1,
             const _Float16* __restrict__ w2p, const float* __restrict__ b2,
             const _Float16* __restrict__ w3p, const float* __restrict__ b3,
             const float* __restrict__ A_ws, float* __restrict__ out) {
    __shared__ __align__(16) _Float16 h1[16][72];
    __shared__ __align__(16) _Float16 h2[16][136];

    const int tid  = threadIdx.x;
    const int lane = tid & 31;
    const int wave = tid >> 5;
    const int m0   = blockIdx.x * 16;

    mlp_front(m0, w1, b1, w2p, b2, h1, h2);

    const _Float16* arow = h2[lane & 15];
    const int koff = (lane & 16) ? 16 : 0;
    const int kb   = (lane & 16) ? 16 : 0;
    const int mrow = (lane & 16) ? 8 : 0;
    v16h af[4];
#pragma unroll
    for (int kc = 0; kc < 4; ++kc) af[kc] = load_afrag(arow + kc * 32 + koff);

    const float inv_nx = 1.f / (float)NX;

    // Each wave owns 8 output channels o; each o spans two 16-column tiles (i halves).
    for (int o = wave; o < COUT; o += 4) {
        float acc[BATCH][8];
#pragma unroll
        for (int b = 0; b < BATCH; ++b)
#pragma unroll
            for (int r = 0; r < 8; ++r) acc[b][r] = 0.f;

#pragma unroll
        for (int half = 0; half < 2; ++half) {
            const int t = o * 2 + half;
            const int n = t * 16 + (lane & 15);       // psi column o*32+i
            const _Float16* wp = w3p + n * 128;
            v8f c = {};
#pragma unroll
            for (int kc = 0; kc < 4; ++kc)
                c = wmma_f16(af[kc], load_bfrag(wp + kc * 32 + kb), c);
            const float bias = b3[n];
            const int   i    = half * 16 + (lane & 15);
#pragma unroll
            for (int b = 0; b < BATCH; ++b) {
                float Ab = A_ws[(b * COUT + o) * CIN + i];
#pragma unroll
                for (int r = 0; r < 8; ++r) acc[b][r] += Ab * (c[r] + bias);
            }
        }
        // Butterfly-sum over i (the 16 lanes of each half-wave).
#pragma unroll
        for (int b = 0; b < BATCH; ++b)
#pragma unroll
            for (int r = 0; r < 8; ++r) {
                float v = acc[b][r];
                v += __shfl_xor(v, 1, 16);
                v += __shfl_xor(v, 2, 16);
                v += __shfl_xor(v, 4, 16);
                v += __shfl_xor(v, 8, 16);
                acc[b][r] = v;
            }
        if ((lane & 15) == 0) {                        // lane 0 -> rows 0..7, lane 16 -> 8..15
#pragma unroll
            for (int b = 0; b < BATCH; ++b) {
                float* op = out + ((size_t)b * COUT + o) * NX + m0 + mrow;
                *(float4*)op       = make_float4(acc[b][0] * inv_nx, acc[b][1] * inv_nx,
                                                 acc[b][2] * inv_nx, acc[b][3] * inv_nx);
                *(float4*)(op + 4) = make_float4(acc[b][4] * inv_nx, acc[b][5] * inv_nx,
                                                 acc[b][6] * inv_nx, acc[b][7] * inv_nx);
            }
        }
    }
}

// ---------------------------------------------------------------------------
extern "C" void kernel_launch(void* const* d_in, const int* in_sizes, int n_in,
                              void* d_out, int out_size, void* d_ws, size_t ws_size,
                              hipStream_t stream) {
    const float* x      = (const float*)d_in[0];
    const float* phi_w1 = (const float*)d_in[1];
    const float* phi_b1 = (const float*)d_in[2];
    const float* phi_w2 = (const float*)d_in[3];
    const float* phi_b2 = (const float*)d_in[4];
    const float* phi_w3 = (const float*)d_in[5];
    const float* phi_b3 = (const float*)d_in[6];
    const float* psi_w1 = (const float*)d_in[7];
    const float* psi_b1 = (const float*)d_in[8];
    const float* psi_w2 = (const float*)d_in[9];
    const float* psi_b2 = (const float*)d_in[10];
    const float* psi_w3 = (const float*)d_in[11];
    const float* psi_b3 = (const float*)d_in[12];
    float* outp = (float*)d_out;

    // Workspace layout (bytes): A (16K) | w2p_phi (16K) | w2p_psi (16K)
    //                           | w3p_phi (256K) | w3p_psi (256K)  => ~560 KB
    char* ws = (char*)d_ws;
    float*    A_ws    = (float*)ws;
    _Float16* w2p_phi = (_Float16*)(ws + 16384);
    _Float16* w2p_psi = (_Float16*)(ws + 32768);
    _Float16* w3p_phi = (_Float16*)(ws + 49152);
    _Float16* w3p_psi = (_Float16*)(ws + 311296);

    lr3d_pack<<<256, 256, 0, stream>>>(phi_w2, phi_w3, psi_w2, psi_w3,
                                       w2p_phi, w3p_phi, w2p_psi, w3p_psi, A_ws);

    lr3d_phi_accum<<<NX / 16, 128, 0, stream>>>(x, phi_w1, phi_b1, w2p_phi, phi_b2,
                                                w3p_phi, phi_b3, A_ws);

    lr3d_psi_out<<<NX / 16, 128, 0, stream>>>(psi_w1, psi_b1, w2p_psi, psi_b2,
                                              w3p_psi, psi_b3, A_ws, outp);
}
